// AttentionHead_3324304687884
// MI455X (gfx1250) — matrix operand order repaired and checked by reading, not compiled, not run on hardware
//
#include <hip/hip_runtime.h>

typedef __attribute__((ext_vector_type(16))) _Float16 v16h;
typedef __attribute__((ext_vector_type(8)))  _Float16 v8h;
typedef __attribute__((ext_vector_type(8)))  float    v8f;
typedef __attribute__((ext_vector_type(4)))  float    v4f;

#define WIDTH 768
#define HEAD  64
#define NSEQ  2048
#define BATCH 8
#define ROWS_TOTAL (BATCH * NSEQ)

static __device__ __forceinline__ v8f wmma16(v16h a, v16h b, v8f c) {
    // D = A(16x32 f16) * B(32x16 f16) + C(16x16 f32)
    return __builtin_amdgcn_wmma_f32_16x16x32_f16(
        /*neg_a=*/false, a, /*neg_b=*/false, b,
        /*c_mod=*/(short)0, c, /*reuse_a=*/false, /*reuse_b=*/false);
}

// ---------------------------------------------------------------------------
// Kernel 0: pack Wq/Wk/Wv (f32 [768][64]) into f16 transposed [3][64][768]
// so projection B-operand per-lane reads are 32B contiguous.
// ---------------------------------------------------------------------------
__global__ void pack_weights(const float* __restrict__ Wq,
                             const float* __restrict__ Wk,
                             const float* __restrict__ Wv,
                             _Float16* __restrict__ Wt) {
    int idx = blockIdx.x * blockDim.x + threadIdx.x;
    const int total = 3 * HEAD * WIDTH;
    if (idx >= total) return;
    int w   = idx / (HEAD * WIDTH);
    int rem = idx % (HEAD * WIDTH);
    int n   = rem / WIDTH;
    int k   = rem % WIDTH;
    const float* W = (w == 0) ? Wq : (w == 1) ? Wk : Wv;
    Wt[idx] = (_Float16)W[(size_t)k * HEAD + n];
}

// ---------------------------------------------------------------------------
// Kernel 1: fused QKV projection.  One wave -> 16 rows x 64 cols x {Q,K,V}.
// Q gets the 1/sqrt(HEAD) scale folded in.  V stored transposed [B][64][2048].
// ---------------------------------------------------------------------------
__global__ __launch_bounds__(128) void qkv_kernel(
    const float* __restrict__ x,
    const _Float16* __restrict__ Wt,     // [3][64][768] f16
    const float* __restrict__ bq, const float* __restrict__ bk,
    const float* __restrict__ bv,
    _Float16* __restrict__ Qh,           // [16384][64]
    _Float16* __restrict__ Kh,           // [16384][64]
    _Float16* __restrict__ Vt)           // [8][64][2048]
{
    const int lane = threadIdx.x & 31;
    const int wave = threadIdx.x >> 5;
    const int rowbase = (blockIdx.x * 4 + wave) * 16;
    const int r    = lane & 15;
    const int hi   = lane >> 4;   // 0: K 0-15, 1: K 16-31 (A layout)
    const int koff = hi * 16;

    v8f acc[3][4];
    #pragma unroll
    for (int w = 0; w < 3; ++w)
        #pragma unroll
        for (int t = 0; t < 4; ++t) acc[w][t] = (v8f)0.f;

    const float* xrow = x + (size_t)(rowbase + r) * WIDTH;

    for (int c = 0; c < WIDTH / 32; ++c) {
        const int kb = c * 32;
        // ---- A operand: 16x32 f16 tile of x ----
        v4f f0 = *(const v4f*)(xrow + kb + koff);
        v4f f1 = *(const v4f*)(xrow + kb + koff + 4);
        v4f f2 = *(const v4f*)(xrow + kb + koff + 8);
        v4f f3 = *(const v4f*)(xrow + kb + koff + 12);
        v16h a;
        #pragma unroll
        for (int i = 0; i < 4; ++i) {
            a[i]      = (_Float16)f0[i];
            a[4 + i]  = (_Float16)f1[i];
            a[8 + i]  = (_Float16)f2[i];
            a[12 + i] = (_Float16)f3[i];
        }
        #pragma unroll
        for (int w = 0; w < 3; ++w) {
            #pragma unroll
            for (int t = 0; t < 4; ++t) {
                const int n = t * 16 + r;                // B col = lane%16
                const _Float16* bp =
                    Wt + (size_t)(w * HEAD + n) * WIDTH + kb + hi * 16;
                v16h bm = *(const v16h*)bp;              // 16 contiguous K vals
                acc[w][t] = wmma16(a, bm, acc[w][t]);
            }
        }
    }

    // ---- epilogue: bias, Q-scale, stores (C layout: VGPR j -> row j or j+8)
    const int row_lo = rowbase + hi * 8;
    const float scale = 0.125f;    // 1/sqrt(64)
    #pragma unroll
    for (int t = 0; t < 4; ++t) {
        const int col = t * 16 + r;
        const float bqv = bq[col], bkv = bk[col], bvv = bv[col];
        #pragma unroll
        for (int j = 0; j < 8; ++j) {
            const int row = row_lo + j;
            Qh[(size_t)row * HEAD + col] = (_Float16)((acc[0][t][j] + bqv) * scale);
            Kh[(size_t)row * HEAD + col] = (_Float16)(acc[1][t][j] + bkv);
            const int bb = row >> 11;       // row / 2048
            const int n  = row & 2047;
            Vt[((size_t)bb * HEAD + col) * NSEQ + n] = (_Float16)(acc[2][t][j] + bvv);
        }
    }
}

// ---------------------------------------------------------------------------
// Kernel 2: flash attention.  Wave = 16 queries, block = 4 waves = 64 queries.
// K/V tiles staged into LDS via CDNA5 async global->LDS loads (ASYNCcnt).
// ---------------------------------------------------------------------------
__global__ __launch_bounds__(128) void flash_kernel(
    const _Float16* __restrict__ Qh, const _Float16* __restrict__ Kh,
    const _Float16* __restrict__ Vt, const int* __restrict__ mask,
    float* __restrict__ out)
{
    __shared__ __align__(32) _Float16 Ks[32 * 64];      // [key][h]
    __shared__ __align__(32) _Float16 Vs[64 * 32];      // [h][key]
    __shared__ __align__(32) _Float16 Ps[4][16 * 32];   // per-wave P staging

    const int tid  = threadIdx.x;
    const int lane = tid & 31;
    const int wave = tid >> 5;
    const int r    = lane & 15;
    const int hi   = lane >> 4;

    const int qglobal = blockIdx.x * 64 + wave * 16;    // 0..16383
    const int b  = qglobal >> 11;
    const int q0 = qglobal & 2047;
    const int rowoff = hi * 8;                          // C-layout row offset

    // Per-thread LDS destinations / global sources for async staging.
    const unsigned kdst = (unsigned)(unsigned long long)&Ks[tid * 16];
    const int vh = tid >> 1, vko = (tid & 1) * 16;
    const unsigned vdst = (unsigned)(unsigned long long)&Vs[vh * 32 + vko];
    const _Float16* vbase = Vt + ((size_t)b * HEAD + vh) * NSEQ + vko;

    // ---- load Q A-operands once (two 32-wide h chunks) ----
    const _Float16* qrow = Qh + (size_t)(qglobal + r) * HEAD;
    v16h aQ[2];
    #pragma unroll
    for (int cch = 0; cch < 2; ++cch) {
        const int hb = cch * 32;
        v8h lo = *(const v8h*)(qrow + hb + hi * 16);
        v8h hh = *(const v8h*)(qrow + hb + hi * 16 + 8);
        #pragma unroll
        for (int i = 0; i < 8; ++i) { aQ[cch][i] = lo[i]; aQ[cch][8 + i] = hh[i]; }
    }

    v8f O[4];
    float m[8], l[8];
    #pragma unroll
    for (int t = 0; t < 4; ++t) O[t] = (v8f)0.f;
    #pragma unroll
    for (int j = 0; j < 8; ++j) { m[j] = -1e30f; l[j] = 0.f; }

    const int* mrow = mask + (size_t)(q0 + rowoff) * NSEQ;

    for (int kt = 0; kt < NSEQ / 32; ++kt) {
        const int keybase = kt * 32;

        // ---- async LDS staging: K tile (32x64) + V tile (64x32) ----
        {
            const _Float16* ksrc =
                Kh + (size_t)(b * NSEQ + keybase) * HEAD + tid * 16;
            const _Float16* vsrc = vbase + keybase;
            // INST_OFFSET applies to both LDS and global addresses (ISA 10.x).
            asm volatile("global_load_async_to_lds_b128 %0, %1, off"
                         :: "v"(kdst), "v"(ksrc) : "memory");
            asm volatile("global_load_async_to_lds_b128 %0, %1, off offset:16"
                         :: "v"(kdst), "v"(ksrc) : "memory");
            asm volatile("global_load_async_to_lds_b128 %0, %1, off"
                         :: "v"(vdst), "v"(vsrc) : "memory");
            asm volatile("global_load_async_to_lds_b128 %0, %1, off offset:16"
                         :: "v"(vdst), "v"(vsrc) : "memory");
            if (kt + 1 < NSEQ / 32) {  // prefetch next K tile toward L2
                __builtin_prefetch(
                    Kh + (size_t)(b * NSEQ + keybase + 32) * HEAD + tid * 16, 0, 0);
            }
        }
        asm volatile("s_wait_asynccnt 0" ::: "memory");
        __syncthreads();

        // ---- scores: S = Q * K^T (two 16-key subtiles, K-dim = 64) ----
        v8f S[2];
        #pragma unroll
        for (int s = 0; s < 2; ++s) {
            const int key = s * 16 + r;                  // B col = key
            v8f cacc = (v8f)0.f;
            #pragma unroll
            for (int cch = 0; cch < 2; ++cch) {
                v16h bK = *(const v16h*)(&Ks[key * 64 + cch * 32 + hi * 16]);
                cacc = wmma16(aQ[cch], bK, cacc);
            }
            S[s] = cacc;
        }

        // ---- mask + online softmax (reductions within 16-lane groups) ----
        float p0v[8], p1v[8], alpha[8];
        #pragma unroll
        for (int j = 0; j < 8; ++j) {
            const int* mp = mrow + (size_t)j * NSEQ + keybase;
            float s0 = (mp[r]      != 0) ? S[0][j] : -1e30f;
            float s1 = (mp[16 + r] != 0) ? S[1][j] : -1e30f;
            float mx = fmaxf(s0, s1);
            #pragma unroll
            for (int d = 1; d < 16; d <<= 1) mx = fmaxf(mx, __shfl_xor(mx, d, 32));
            const float mnew = fmaxf(m[j], mx);
            const float a  = __expf(m[j] - mnew);
            const float e0 = __expf(s0 - mnew);
            const float e1 = __expf(s1 - mnew);
            float rs = e0 + e1;
            #pragma unroll
            for (int d = 1; d < 16; d <<= 1) rs += __shfl_xor(rs, d, 32);
            l[j] = l[j] * a + rs;
            m[j] = mnew;
            alpha[j] = a;
            p0v[j] = e0; p1v[j] = e1;
        }
        #pragma unroll
        for (int t = 0; t < 4; ++t)
            #pragma unroll
            for (int j = 0; j < 8; ++j) O[t][j] *= alpha[j];

        // ---- stage P (C layout) -> LDS -> reload in A layout ----
        _Float16* Pw = &Ps[wave][0];
        #pragma unroll
        for (int j = 0; j < 8; ++j) {
            Pw[(rowoff + j) * 32 + r]      = (_Float16)p0v[j];
            Pw[(rowoff + j) * 32 + 16 + r] = (_Float16)p1v[j];
        }
        asm volatile("s_wait_dscnt 0" ::: "memory");  // wave-local LDS RAW

        v16h aP;
        {
            const _Float16* pr = Pw + r * 32;
            v8h lo = *(const v8h*)(pr + hi * 16);
            v8h hh = *(const v8h*)(pr + hi * 16 + 8);
            #pragma unroll
            for (int i = 0; i < 8; ++i) { aP[i] = lo[i]; aP[8 + i] = hh[i]; }
        }

        // ---- O += P * V  (4 h-tiles of 16) ----
        #pragma unroll
        for (int t = 0; t < 4; ++t) {
            const int h = t * 16 + r;                    // B col = h
            v16h bV = *(const v16h*)(&Vs[h * 32 + hi * 16]);
            O[t] = wmma16(aP, bV, O[t]);
        }

        __syncthreads();
    }

    // ---- epilogue: normalize and store fp32 ----
    #pragma unroll
    for (int j = 0; j < 8; ++j) {
        const float inv = 1.0f / l[j];
        #pragma unroll
        for (int t = 0; t < 4; ++t) {
            out[(size_t)(qglobal + rowoff + j) * HEAD + t * 16 + r] =
                O[t][j] * inv;
        }
    }
}

// ---------------------------------------------------------------------------
extern "C" void kernel_launch(void* const* d_in, const int* in_sizes, int n_in,
                              void* d_out, int out_size, void* d_ws, size_t ws_size,
                              hipStream_t stream) {
    const float* x  = (const float*)d_in[0];
    const float* Wq = (const float*)d_in[1];
    const float* bq = (const float*)d_in[2];
    const float* Wk = (const float*)d_in[3];
    const float* bk = (const float*)d_in[4];
    const float* Wv = (const float*)d_in[5];
    const float* bv = (const float*)d_in[6];
    const int* mask = (const int*)d_in[7];
    float* out = (float*)d_out;

    char* ws = (char*)d_ws;
    const size_t seg = (size_t)ROWS_TOTAL * HEAD * sizeof(_Float16); // 2 MB
    _Float16* Qh = (_Float16*)(ws);
    _Float16* Kh = (_Float16*)(ws + seg);
    _Float16* Vt = (_Float16*)(ws + 2 * seg);
    _Float16* Wt = (_Float16*)(ws + 3 * seg);   // 3*64*768 f16 = 288 KB

    pack_weights<<<(3 * HEAD * WIDTH + 255) / 256, 256, 0, stream>>>(Wq, Wk, Wv, Wt);
    qkv_kernel<<<ROWS_TOTAL / 64, 128, 0, stream>>>(x, Wt, bq, bk, bv, Qh, Kh, Vt);
    flash_kernel<<<ROWS_TOTAL / 64, 128, 0, stream>>>(Qh, Kh, Vt, mask, out);
}
